// GroupedAttention_3633542332620
// MI455X (gfx1250) — matrix operand order repaired and checked by reading, not compiled, not run on hardware
//
#include <hip/hip_runtime.h>
#include <stdint.h>

// ---------------------------------------------------------------------------
// Grouped attention for MI455X (gfx1250), wave32 + v_wmma_f32_16x16x32_bf16.
// Pipeline: prep_w (weight transpose->bf16)  ->  qkv_gemm (block-diag GEMM)
//           -> attn_fa (flash attention, S^T trick) -> proj_gemm (+bias).
// This revision: use native __bf16 casts so clang emits v_cvt_pk_bf16_f32
// instead of the 3-VALU-op manual RNE sequence.
// ---------------------------------------------------------------------------

typedef __attribute__((ext_vector_type(16))) __bf16          v16bf;
typedef __attribute__((ext_vector_type(8)))  float           v8f;

#define BATCH  2
#define SEQ    2048
#define CDIM   1024
#define HEADS  16
#define HD     64
#define NTOK   (BATCH * SEQ)

union U8x  { uint4  u;  unsigned short s[8]; };
union F8x  { float4 v[2]; float f[8]; };
union BfU  { __bf16 b; unsigned short s; };

__device__ __forceinline__ unsigned short f2bf(float x) {
  BfU u; u.b = (__bf16)x;            // native cvt (v_cvt_pk_bf16_f32 pairs)
  return u.s;
}
__device__ __forceinline__ __bf16 us2bf(unsigned short x) {
  BfU u; u.s = x; return u.b;
}

__device__ __forceinline__ v8f wmma_bf16(v16bf a, v16bf b, v8f c) {
  // (neg_a, A, neg_b, B, c_mod, C, reuse_a, reuse_b)
  return __builtin_amdgcn_wmma_f32_16x16x32_bf16(false, a, false, b,
                                                 (short)0, c, false, false);
}

// ---------------------------------------------------------------------------
// Kernel 0: weights -> bf16, transposed to [g][out][in] so B-fragments are
// contiguous-K per lane.
// ---------------------------------------------------------------------------
__global__ void prep_w(const float* __restrict__ wqkv, const float* __restrict__ wproj,
                       unsigned short* __restrict__ wqkvT, unsigned short* __restrict__ wprojT) {
  int idx = blockIdx.x * 256 + threadIdx.x;
  const int n1 = 8 * 384 * 128;                 // qkv elements
  if (idx < n1) {
    int g = idx / (384 * 128);
    int r = idx - g * 384 * 128;
    int o = r >> 7;                             // out channel within group
    int i = r & 127;                            // in channel within group
    wqkvT[idx] = f2bf(wqkv[(size_t)(g * 128 + i) * 384 + o]);
  } else if (idx < n1 + 8 * 128 * 128) {
    int j = idx - n1;
    int g = j >> 14;
    int r = j & 16383;
    int o = r >> 7;
    int i = r & 127;
    wprojT[j] = f2bf(wproj[(size_t)(g * 128 + i) * 128 + o]);
  }
}

// ---------------------------------------------------------------------------
// Kernel 1: QKV block-diagonal GEMM. One wave -> one 16x16 output tile.
// Writes q (scaled by 1/8) and k as [B,H,N,D] bf16, v transposed [B,H,D,N].
// ---------------------------------------------------------------------------
__global__ void qkv_gemm(const float* __restrict__ x, const unsigned short* __restrict__ wT,
                         unsigned short* __restrict__ q, unsigned short* __restrict__ k,
                         unsigned short* __restrict__ vT) {
  int gwid = (blockIdx.x * 256 + threadIdx.x) >> 5;   // 256*192 waves
  int lane = threadIdx.x & 31;
  int hh = lane >> 4, ln = lane & 15;
  int mt = gwid / 192, nt = gwid - mt * 192;
  int n0 = nt * 16;                 // global qkv channel base (0..3071)
  int g  = n0 / 384;                // group
  int o0 = n0 - g * 384;            // column within group
  int tok0 = mt * 16;

  const float*          xrow = x  + (size_t)(tok0 + ln) * CDIM + g * 128;
  const unsigned short* wcol = wT + ((size_t)g * 384 + o0 + ln) * 128;

  v8f acc = {};
#pragma unroll
  for (int kb = 0; kb < 128; kb += 32) {
    // A-frag (x, f32->bf16): chunks at K = kb+8h and kb+8h+16
    F8x a0, a1;
    const float* pa = xrow + kb + 8 * hh;
    a0.v[0] = *(const float4*)(pa);      a0.v[1] = *(const float4*)(pa + 4);
    a1.v[0] = *(const float4*)(pa + 16); a1.v[1] = *(const float4*)(pa + 20);
    v16bf af;
#pragma unroll
    for (int e = 0; e < 8; ++e) { af[e] = (__bf16)a0.f[e]; af[e + 8] = (__bf16)a1.f[e]; }
    // B-frag (weights): chunks at K = kb+8h and kb+8h+16
    U8x b0, b1;
    const unsigned short* pb = wcol + kb + 8 * hh;
    b0.u = *(const uint4*)(pb); b1.u = *(const uint4*)(pb + 16);
    v16bf bfr;
#pragma unroll
    for (int e = 0; e < 8; ++e) { bfr[e] = us2bf(b0.s[e]); bfr[e + 8] = us2bf(b1.s[e]); }
    acc = wmma_bf16(af, bfr, acc);
  }

  int f     = n0 + ln;           // flat qkv channel
  int three = f >> 10;           // 0=q 1=k 2=v (wave-uniform: 16 | 1024)
  int rem   = f & 1023;
  int head  = rem >> 6;          // wave-uniform (16 | 64)
  int d     = rem & 63;
#pragma unroll
  for (int r = 0; r < 8; ++r) {
    int token = tok0 + r + 8 * hh;           // C-layout row
    int b = token >> 11, n = token & 2047;
    int bh = b * HEADS + head;
    float val = acc[r];
    if (three == 0)      q [((size_t)bh * SEQ + n) * HD + d] = f2bf(val * 0.125f);
    else if (three == 1) k [((size_t)bh * SEQ + n) * HD + d] = f2bf(val);
    else                 vT[((size_t)bh * HD  + d) * SEQ + n] = f2bf(val);
  }
}

// ---------------------------------------------------------------------------
// Kernel 2: flash attention. One wave handles 16 query rows of one (b,h).
// Computes S^T = K x Q^T so each lane owns one query column: online-softmax
// stats are per-lane scalars (one shfl_xor(16) to merge the lane pair).
// Then out^T = V^T x P^T accumulated in 4 f32 WMMA tiles (d = 0..63).
// ---------------------------------------------------------------------------
__global__ void attn_fa(const unsigned short* __restrict__ q, const unsigned short* __restrict__ k,
                        const unsigned short* __restrict__ vT, unsigned short* __restrict__ ao) {
  int gwid = (blockIdx.x * 256 + threadIdx.x) >> 5;   // 32*128 waves
  int lane = threadIdx.x & 31;
  int hh = lane >> 4, ln = lane & 15;
  int ntile = gwid & 127;
  int bh = gwid >> 7;
  int b = bh >> 4, h = bh & 15;

  const unsigned short* qb  = q  + (size_t)bh * SEQ * HD;
  const unsigned short* kbp = k  + (size_t)bh * SEQ * HD;
  const unsigned short* vb  = vT + (size_t)bh * HD * SEQ;

  // Q^T B-frags (K = d, two chunks of 32); lane = query column, d chunks at 8h and 8h+16.
  v16bf qf[2];
  int qrow = ntile * 16 + ln;
#pragma unroll
  for (int f = 0; f < 2; ++f) {
    U8x c0, c1;
    const unsigned short* p = qb + (size_t)qrow * HD + f * 32 + 8 * hh;
    c0.u = *(const uint4*)p; c1.u = *(const uint4*)(p + 16);
    v16bf fr;
#pragma unroll
    for (int e = 0; e < 8; ++e) { fr[e] = us2bf(c0.s[e]); fr[e + 8] = us2bf(c1.s[e]); }
    qf[f] = fr;
  }

  v8f acc[4];
#pragma unroll
  for (int t = 0; t < 4; ++t) acc[t] = (v8f){};
  float mrun = -__builtin_inff();
  float lrun = 0.f;

  for (int mb = 0; mb < SEQ; mb += 32) {
    // K A-frags: two key tiles (m rows), two d-chunks each
    v16bf ka[2][2];
#pragma unroll
    for (int mt = 0; mt < 2; ++mt) {
      const unsigned short* p = kbp + (size_t)(mb + mt * 16 + ln) * HD;
#pragma unroll
      for (int f = 0; f < 2; ++f) {
        U8x c0, c1;
        const unsigned short* pp = p + f * 32 + 8 * hh;
        c0.u = *(const uint4*)pp; c1.u = *(const uint4*)(pp + 16);
        v16bf fr;
#pragma unroll
        for (int e = 0; e < 8; ++e) { fr[e] = us2bf(c0.s[e]); fr[e + 8] = us2bf(c1.s[e]); }
        ka[mt][f] = fr;
      }
    }
    // S^T tiles: rows m, cols n (queries). Lane owns column n = ln.
    v8f s0 = {}, s1 = {};
    s0 = wmma_bf16(ka[0][0], qf[0], s0);
    s0 = wmma_bf16(ka[0][1], qf[1], s0);
    s1 = wmma_bf16(ka[1][0], qf[0], s1);
    s1 = wmma_bf16(ka[1][1], qf[1], s1);

    // Online softmax (per query column; merge lane pair l <-> l+16)
    float cmax = -__builtin_inff();
#pragma unroll
    for (int r = 0; r < 8; ++r) cmax = fmaxf(cmax, fmaxf(s0[r], s1[r]));
    cmax = fmaxf(cmax, __shfl_xor(cmax, 16, 32));
    float mnew  = fmaxf(mrun, cmax);
    float alpha = __expf(mrun - mnew);
    float p0[8], p1[8]; float psum = 0.f;
#pragma unroll
    for (int r = 0; r < 8; ++r) {
      p0[r] = __expf(s0[r] - mnew);
      p1[r] = __expf(s1[r] - mnew);
      psum += p0[r] + p1[r];
    }
    psum += __shfl_xor(psum, 16, 32);
    lrun = lrun * alpha + psum;
    mrun = mnew;
#pragma unroll
    for (int t = 0; t < 4; ++t)
#pragma unroll
      for (int r = 0; r < 8; ++r) acc[t][r] *= alpha;

    // Repack P^T (C-layout) into a B-frag: K = m offset = 16h + e
    v16bf pf;
#pragma unroll
    for (int r = 0; r < 8; ++r) {
      float t0 = __shfl_xor(p0[r], 16, 32);
      float t1 = __shfl_xor(p1[r], 16, 32);
      pf[r]     = (__bf16)(hh ? t1    : p0[r]);   // m = 16h + r
      pf[8 + r] = (__bf16)(hh ? p1[r] : t0);      // m = 16h + 8 + r
    }

    // out^T += V^T x P^T : 4 d-tiles, A-frag contiguous from [D,N] layout
#pragma unroll
    for (int dt = 0; dt < 4; ++dt) {
      const unsigned short* p = vb + (size_t)(dt * 16 + ln) * SEQ + mb + 16 * hh;
      U8x c0, c1;
      c0.u = *(const uint4*)p; c1.u = *(const uint4*)(p + 8);
      v16bf fr;
#pragma unroll
      for (int e = 0; e < 8; ++e) { fr[e] = us2bf(c0.s[e]); fr[e + 8] = us2bf(c1.s[e]); }
      acc[dt] = wmma_bf16(fr, pf, acc[dt]);
    }
  }

  float inv = 1.f / lrun;
  int n = ntile * 16 + ln;
  unsigned short* orow = ao + ((size_t)b * SEQ + n) * CDIM + h * HD;
#pragma unroll
  for (int dt = 0; dt < 4; ++dt)
#pragma unroll
    for (int r = 0; r < 8; ++r) {
      int d = dt * 16 + r + 8 * hh;            // C-layout row index = d
      orow[d] = f2bf(acc[dt][r] * inv);
    }
}

// ---------------------------------------------------------------------------
// Kernel 3: output projection (block-diagonal, 128->128 per group) + bias.
// ---------------------------------------------------------------------------
__global__ void proj_gemm(const unsigned short* __restrict__ ain, const unsigned short* __restrict__ wT,
                          const float* __restrict__ bias, float* __restrict__ out) {
  int gwid = (blockIdx.x * 256 + threadIdx.x) >> 5;   // 256*64 waves
  int lane = threadIdx.x & 31;
  int hh = lane >> 4, ln = lane & 15;
  int mt = gwid >> 6, nt = gwid & 63;
  int n0 = nt * 16;
  int g  = n0 >> 7;
  int o0 = n0 & 127;
  int tok0 = mt * 16;

  const unsigned short* arow = ain + (size_t)(tok0 + ln) * CDIM + g * 128;
  const unsigned short* wcol = wT + ((size_t)g * 128 + o0 + ln) * 128;

  v8f acc = {};
#pragma unroll
  for (int kb = 0; kb < 128; kb += 32) {
    U8x a0, a1, b0, b1;
    const unsigned short* pa = arow + kb + 8 * hh;
    a0.u = *(const uint4*)pa; a1.u = *(const uint4*)(pa + 16);
    const unsigned short* pb = wcol + kb + 8 * hh;
    b0.u = *(const uint4*)pb; b1.u = *(const uint4*)(pb + 16);
    v16bf af, bfr;
#pragma unroll
    for (int e = 0; e < 8; ++e) {
      af[e]  = us2bf(a0.s[e]); af[e + 8]  = us2bf(a1.s[e]);
      bfr[e] = us2bf(b0.s[e]); bfr[e + 8] = us2bf(b1.s[e]);
    }
    acc = wmma_bf16(af, bfr, acc);
  }
  float bv = bias[n0 + ln];
#pragma unroll
  for (int r = 0; r < 8; ++r) {
    int token = tok0 + r + 8 * hh;
    out[(size_t)token * CDIM + n0 + ln] = acc[r] + bv;
  }
}

// ---------------------------------------------------------------------------
extern "C" void kernel_launch(void* const* d_in, const int* in_sizes, int n_in,
                              void* d_out, int out_size, void* d_ws, size_t ws_size,
                              hipStream_t stream) {
  (void)in_sizes; (void)n_in; (void)out_size; (void)ws_size;
  const float* x     = (const float*)d_in[0];
  const float* wqkv  = (const float*)d_in[1];
  const float* wproj = (const float*)d_in[2];
  const float* bproj = (const float*)d_in[3];
  float* out = (float*)d_out;

  // Workspace carve-up (bf16/ushort elements), ~33 MB total.
  unsigned short* ws     = (unsigned short*)d_ws;
  unsigned short* q      = ws;
  unsigned short* k      = q  + (size_t)BATCH * HEADS * SEQ * HD;
  unsigned short* vT     = k  + (size_t)BATCH * HEADS * SEQ * HD;
  unsigned short* ao     = vT + (size_t)BATCH * HEADS * SEQ * HD;
  unsigned short* wqkvT  = ao + (size_t)NTOK * CDIM;
  unsigned short* wprojT = wqkvT + (size_t)8 * 384 * 128;

  prep_w   <<<(8 * 384 * 128 + 8 * 128 * 128 + 255) / 256, 256, 0, stream>>>(wqkv, wproj, wqkvT, wprojT);
  qkv_gemm <<<(256 * 192) / 8, 256, 0, stream>>>(x, wqkvT, q, k, vT);   // 49152 waves
  attn_fa  <<<(32 * 128) / 8,  256, 0, stream>>>(q, k, vT, ao);         // 4096 waves
  proj_gemm<<<(256 * 64) / 8,  256, 0, stream>>>(ao, wprojT, bproj, out); // 16384 waves
}
